// DualEncoderPerformer_86766929314198
// MI455X (gfx1250) — matrix operand order repaired and checked by reading, not compiled, not run on hardware
//
#include <hip/hip_runtime.h>
#include <hip/hip_bf16.h>

#define N_ROWS 16384
#define DIM    256
#define MARGIN 0.3f
#define EPS    1e-8f
#define TILE   128
#define KCH    32
#define NCHUNK (DIM / KCH)

typedef __attribute__((ext_vector_type(16))) _Float16 v16h;
typedef __attribute__((ext_vector_type(8)))  _Float16 v8h;
typedef __attribute__((ext_vector_type(8)))  float    v8f;
typedef __attribute__((ext_vector_type(4)))  float    v4f;
typedef __attribute__((ext_vector_type(4)))  int      v4i;

// ---- async global->LDS (CDNA5), with guaranteed-compiling fallback -----
#if defined(__has_builtin)
#if __has_builtin(__builtin_amdgcn_global_load_async_to_lds_b128)
#define HAS_ASYNC_LDS 1
#endif
#endif
#ifndef HAS_ASYNC_LDS
#define HAS_ASYNC_LDS 0
#endif

#if HAS_ASYNC_LDS
typedef __attribute__((address_space(1))) v4i* gv4i_p;   // global <4 x i32>*
typedef __attribute__((address_space(3))) v4i* lv4i_p;   // LDS    <4 x i32>*
#endif

__device__ __forceinline__ void cp16_to_lds(_Float16* lds, const _Float16* g) {
#if HAS_ASYNC_LDS
    // generic->as(1) is numerically identical; generic LDS addr low 32 bits
    // are the LDS offset (ISA 10.2), so integer casts are safe here.
    __builtin_amdgcn_global_load_async_to_lds_b128(
        (gv4i_p)(unsigned long long)g,
        (lv4i_p)(unsigned int)(unsigned long long)lds,
        0, 0);
#else
    *(v8h*)lds = *(const v8h*)g;
#endif
}

__device__ __forceinline__ void async_wait() {
#if HAS_ASYNC_LDS
    asm volatile("s_wait_asynccnt 0x0" ::: "memory");
#endif
}

// ---------------------------------------------------------------- zero
__global__ void zero_f32(float* __restrict__ p, int n) {
    int i = blockIdx.x * 256 + threadIdx.x;
    if (i < n) p[i] = 0.0f;
}

// ------------------------------------------------- row-normalize -> f16
// One wave per row; rows [0,N) come from x -> xh, rows [N,2N) from y -> yh.
__global__ void normalize_rows(const float* __restrict__ x,
                               const float* __restrict__ y,
                               _Float16* __restrict__ xh,
                               _Float16* __restrict__ yh) {
    const int w    = threadIdx.x >> 5;
    const int lane = threadIdx.x & 31;
    int rowAll = blockIdx.x * 8 + w;             // 0 .. 2N-1
    const float* src; _Float16* dst; int r;
    if (rowAll < N_ROWS) { src = x; dst = xh; r = rowAll; }
    else                 { src = y; dst = yh; r = rowAll - N_ROWS; }

    const float* p = src + (size_t)r * DIM + lane * 8;
    v4f a = *(const v4f*)p;
    v4f b = *(const v4f*)(p + 4);
    float ss = a.x*a.x + a.y*a.y + a.z*a.z + a.w*a.w
             + b.x*b.x + b.y*b.y + b.z*b.z + b.w*b.w;
#pragma unroll
    for (int m = 16; m >= 1; m >>= 1) ss += __shfl_xor(ss, m, 32);
    float inv = 1.0f / fmaxf(sqrtf(ss), EPS);

    v8h h;
    h[0] = (_Float16)(a.x * inv); h[1] = (_Float16)(a.y * inv);
    h[2] = (_Float16)(a.z * inv); h[3] = (_Float16)(a.w * inv);
    h[4] = (_Float16)(b.x * inv); h[5] = (_Float16)(b.y * inv);
    h[6] = (_Float16)(b.z * inv); h[7] = (_Float16)(b.w * inv);
    *(v8h*)(dst + (size_t)r * DIM + lane * 8) = h;
}

// --------------------------------------------- exact fp32 diagonal term
__global__ void diag_kernel(const float* __restrict__ x,
                            const float* __restrict__ y,
                            float* __restrict__ diag) {
    const int w    = threadIdx.x >> 5;
    const int lane = threadIdx.x & 31;
    const int r = blockIdx.x * 8 + w;
    const float* px = x + (size_t)r * DIM + lane * 8;
    const float* py = y + (size_t)r * DIM + lane * 8;
    float dot = 0.f, sx = 0.f, sy = 0.f;
#pragma unroll
    for (int t = 0; t < 8; ++t) {
        float xv = px[t], yv = py[t];
        dot += xv * yv; sx += xv * xv; sy += yv * yv;
    }
#pragma unroll
    for (int m = 16; m >= 1; m >>= 1) {
        dot += __shfl_xor(dot, m, 32);
        sx  += __shfl_xor(sx,  m, 32);
        sy  += __shfl_xor(sy,  m, 32);
    }
    if (lane == 0)
        diag[r] = dot / (fmaxf(sqrtf(sx), EPS) * fmaxf(sqrtf(sy), EPS));
}

// ------------- fused WMMA GEMM tile + exp + row/col exp-sum reduction
// Double-buffered K chunks staged with async global->LDS copies.
__global__ void __launch_bounds__(256)
sim_tile_kernel(const _Float16* __restrict__ xh,
                const _Float16* __restrict__ yh,
                float* __restrict__ rowsum,
                float* __restrict__ colsum) {
    __shared__ __align__(16) _Float16 As[2][TILE * KCH];   // 2 x 8 KB
    __shared__ __align__(16) _Float16 Bs[2][TILE * KCH];   // 2 x 8 KB
    __shared__ float rs[TILE];
    __shared__ float cs[TILE];

    const int tid  = threadIdx.x;
    const int w    = tid >> 5;
    const int lane = tid & 31;
    const int l16  = lane & 15;
    const int hi   = lane >> 4;                 // 0 or 1 (half-wave)
    const int i0 = blockIdx.x * TILE;           // rows of x  -> rowsum
    const int j0 = blockIdx.y * TILE;           // rows of y  -> colsum
    const int wtRow = (w & 3) * 32;             // wave's 32-row block
    const int wtCol = (w >> 2) * 64;            // wave's 64-col block

    // this thread's two staging vectors per matrix per chunk
    const int r0 = (0 * 256 + tid) >> 2, s0 = ((0 * 256 + tid) & 3) * 8;
    const int r1 = (1 * 256 + tid) >> 2, s1 = ((1 * 256 + tid) & 3) * 8;

    if (tid < TILE) { rs[tid] = 0.0f; cs[tid] = 0.0f; }

    v8f acc[2][4];
    const v8f vzero = {0.f,0.f,0.f,0.f,0.f,0.f,0.f,0.f};
#pragma unroll
    for (int tm = 0; tm < 2; ++tm)
#pragma unroll
        for (int tn = 0; tn < 4; ++tn) acc[tm][tn] = vzero;

    // prologue: stage chunk 0 into buffer 0
    {
        cp16_to_lds(&As[0][r0 * KCH + s0], &xh[(size_t)(i0 + r0) * DIM + s0]);
        cp16_to_lds(&Bs[0][r0 * KCH + s0], &yh[(size_t)(j0 + r0) * DIM + s0]);
        cp16_to_lds(&As[0][r1 * KCH + s1], &xh[(size_t)(i0 + r1) * DIM + s1]);
        cp16_to_lds(&Bs[0][r1 * KCH + s1], &yh[(size_t)(j0 + r1) * DIM + s1]);
    }

    for (int kc = 0; kc < NCHUNK; ++kc) {
        const int cur = kc & 1;
        async_wait();          // own async copies for buf[cur] complete
        __syncthreads();       // all waves' copies visible; prev compute done

        if (kc + 1 < NCHUNK) { // prefetch next chunk into the other buffer
            const int nxt = cur ^ 1;
            const int kk  = (kc + 1) * KCH;
            cp16_to_lds(&As[nxt][r0 * KCH + s0], &xh[(size_t)(i0 + r0) * DIM + kk + s0]);
            cp16_to_lds(&Bs[nxt][r0 * KCH + s0], &yh[(size_t)(j0 + r0) * DIM + kk + s0]);
            cp16_to_lds(&As[nxt][r1 * KCH + s1], &xh[(size_t)(i0 + r1) * DIM + kk + s1]);
            cp16_to_lds(&Bs[nxt][r1 * KCH + s1], &yh[(size_t)(j0 + r1) * DIM + kk + s1]);
        }

        // one K=32 WMMA step from buf[cur]
        v16h afr[2], bfr[4];
#pragma unroll
        for (int tm = 0; tm < 2; ++tm) {
            int M = wtRow + tm * 16 + l16;
            v8h lo = *(const v8h*)&As[cur][M * KCH + hi * 8];
            v8h hh = *(const v8h*)&As[cur][M * KCH + 16 + hi * 8];
#pragma unroll
            for (int t = 0; t < 8; ++t) { afr[tm][t] = lo[t]; afr[tm][8 + t] = hh[t]; }
        }
#pragma unroll
        for (int tn = 0; tn < 4; ++tn) {
            int Nl = wtCol + tn * 16 + l16;
            v8h lo = *(const v8h*)&Bs[cur][Nl * KCH + hi * 8];
            v8h hh = *(const v8h*)&Bs[cur][Nl * KCH + 16 + hi * 8];
#pragma unroll
            for (int t = 0; t < 8; ++t) { bfr[tn][t] = lo[t]; bfr[tn][8 + t] = hh[t]; }
        }
#pragma unroll
        for (int tm = 0; tm < 2; ++tm)
#pragma unroll
            for (int tn = 0; tn < 4; ++tn)
                acc[tm][tn] = __builtin_amdgcn_wmma_f32_16x16x32_f16(
                    false, afr[tm], false, bfr[tn],
                    (short)0, acc[tm][tn], false, false);
    }

    // exp() in registers, then column partials (both half-waves hit the
    // same cs[N] -> only 2-way LDS atomic conflict)
#pragma unroll
    for (int tn = 0; tn < 4; ++tn) {
        float cp = 0.0f;
#pragma unroll
        for (int tm = 0; tm < 2; ++tm)
#pragma unroll
            for (int r = 0; r < 8; ++r) {
                float ev = __expf(acc[tm][tn][r]);
                acc[tm][tn][r] = ev;
                cp += ev;
            }
        atomicAdd(&cs[wtCol + tn * 16 + l16], cp);
    }
    // row partials: fold across the 4 column tiles in-lane first
#pragma unroll
    for (int tm = 0; tm < 2; ++tm)
#pragma unroll
        for (int r = 0; r < 8; ++r) {
            float rp = acc[tm][0][r] + acc[tm][1][r] + acc[tm][2][r] + acc[tm][3][r];
            atomicAdd(&rs[wtRow + tm * 16 + hi * 8 + r], rp);
        }

    __syncthreads();
    if (tid < TILE) {
        atomicAdd(&rowsum[i0 + tid], rs[tid]);
        atomicAdd(&colsum[j0 + tid], cs[tid]);
    }
}

// --------------------------------------------------- final scalar loss
__global__ void final_reduce(const float* __restrict__ rowsum,
                             const float* __restrict__ colsum,
                             const float* __restrict__ diag,
                             float* __restrict__ out) {
    __shared__ float red[8];
    float acc = 0.0f;
    for (int i = threadIdx.x; i < N_ROWS; i += 256) {
        float d  = diag[i];
        float m  = __expf(d - MARGIN);
        float ed = __expf(d);
        acc += m / (m + (rowsum[i] - ed));   // x -> y direction
        acc += m / (m + (colsum[i] - ed));   // y -> x direction
    }
#pragma unroll
    for (int msk = 16; msk >= 1; msk >>= 1) acc += __shfl_xor(acc, msk, 32);
    const int w = threadIdx.x >> 5, lane = threadIdx.x & 31;
    if (lane == 0) red[w] = acc;
    __syncthreads();
    if (threadIdx.x == 0) {
        float t = 0.0f;
#pragma unroll
        for (int k = 0; k < 8; ++k) t += red[k];
        out[0] = -t / (float)N_ROWS;
    }
}

// ---------------------------------------------------------------- launch
extern "C" void kernel_launch(void* const* d_in, const int* in_sizes, int n_in,
                              void* d_out, int out_size, void* d_ws, size_t ws_size,
                              hipStream_t stream) {
    const float* x = (const float*)d_in[0];
    const float* y = (const float*)d_in[1];
    float* out = (float*)d_out;

    char* ws = (char*)d_ws;
    _Float16* xh = (_Float16*)ws;                                   // 8 MB
    _Float16* yh = (_Float16*)(ws + (size_t)N_ROWS * DIM * 2);      // 8 MB
    float* rowsum = (float*)(ws + (size_t)N_ROWS * DIM * 4);        // 64 KB
    float* colsum = rowsum + N_ROWS;                                // 64 KB
    float* diag   = colsum + N_ROWS;                                // 64 KB

    zero_f32<<<(2 * N_ROWS + 255) / 256, 256, 0, stream>>>(rowsum, 2 * N_ROWS);
    normalize_rows<<<(2 * N_ROWS) / 8, 256, 0, stream>>>(x, y, xh, yh);
    diag_kernel<<<N_ROWS / 8, 256, 0, stream>>>(x, y, diag);
    dim3 grid(N_ROWS / TILE, N_ROWS / TILE);
    sim_tile_kernel<<<grid, 256, 0, stream>>>(xh, yh, rowsum, colsum);
    final_reduce<<<1, 256, 0, stream>>>(rowsum, colsum, diag, out);
}
